// GeometricAttentionLayer_84482006712467
// MI455X (gfx1250) — matrix operand order repaired and checked by reading, not compiled, not run on hardware
//
#include <hip/hip_runtime.h>
#include <hip/hip_bf16.h>

// ---- CDNA5 WMMA / TDM types ----
typedef __attribute__((ext_vector_type(16))) __bf16  v16bf;
typedef __attribute__((ext_vector_type(8)))  __bf16  v8bf;
typedef __attribute__((ext_vector_type(8)))  float   v8f;
typedef __attribute__((ext_vector_type(4)))  unsigned u32x4;
typedef __attribute__((ext_vector_type(8)))  int      i32x8;
typedef __attribute__((ext_vector_type(4)))  int      i32x4;

#define C_IN 256   // K for all projections
#define HD   256   // H*D
#define HEADS 4
#define DHEAD 64
#define STRIP 64   // output columns per block (4 x 16-wide WMMA tiles)

__device__ __forceinline__ void store_out(float* p, float v)  { *p = v; }
__device__ __forceinline__ void store_out(__bf16* p, float v) { *p = (__bf16)v; }

// ------------------------------------------------------------------
// GEMM: C[M,ncols] = A_bf16[M,256] @ W^T_bf16[ncols,256]^T + bias
// Block = 4 waves = 4 row tiles sharing ONE 64-column strip of W^T.
// The 64x256 bf16 strip (32 KB) is staged into LDS once per block via
// the Tensor Data Mover (tensor_load_to_lds + s_wait_tensorcnt), then
// all WMMA B operands come from ds_load_b128 while A streams from
// global. Each wave: 16x64 strip, 4 accumulators, A fragment reused.
// ------------------------------------------------------------------
template <typename OutT>
__global__ __launch_bounds__(128)
void gemm_bf16_wmma(const __bf16* __restrict__ A,
                    const __bf16* __restrict__ Wt,   // [ncols, 256] row-major
                    const float*  __restrict__ bias, // [ncols]
                    OutT* __restrict__ C,
                    int M, int ncols)
{
    __shared__ __bf16 lwt[STRIP * C_IN];   // 32 KB weight strip

    const int ngroups = ncols / STRIP;
    const int rb = blockIdx.x / ngroups;   // row-block (4 row tiles)
    const int ng = blockIdx.x % ngroups;   // 64-column strip index
    const int wv = threadIdx.x >> 5;
    const int lane = threadIdx.x & 31;

    // ---- stage W^T[ng*64 .. ng*64+63, :] into LDS ----
#if __has_builtin(__builtin_amdgcn_tensor_load_to_lds) && \
    __has_builtin(__builtin_amdgcn_s_wait_tensorcnt)
    if (wv == 0) {                                    // one wave issues the TDM op
        const unsigned lds_off = (unsigned)(size_t)&lwt[0];
        const unsigned long long ga =
            (unsigned long long)(size_t)(Wt + (size_t)ng * STRIP * C_IN);
        // D# group 0: count=1 | lds_addr | global_addr[56:0] | type=2
        u32x4 g0;
        g0[0] = 1u;
        g0[1] = lds_off;
        g0[2] = (unsigned)(ga & 0xFFFFFFFFu);
        g0[3] = (unsigned)((ga >> 32) & 0x01FFFFFFu) | (2u << 30);
        // D# group 1: data_size=1 (2B elems); tensor 256 x 64; tile 256 x 64;
        // dim0 stride = 256 elements
        i32x8 g1;
        g1[0] = (1 << 16);                 // data_size=1, no mask/pad/iterate
        g1[1] = (C_IN & 0xFFFF) << 16;     // tensor_dim0 lo16
        g1[2] = (STRIP & 0xFFFF) << 16;    // tensor_dim0 hi16=0 | tensor_dim1 lo16
        g1[3] = (C_IN & 0xFFFF) << 16;     // tensor_dim1 hi16=0 | tile_dim0
        g1[4] = (STRIP & 0xFFFF);          // tile_dim1 | tile_dim2=0
        g1[5] = C_IN;                      // tensor_dim0_stride lo32
        g1[6] = 0;                         // stride hi | dim1_stride lo (unused, 2D)
        g1[7] = 0;
        i32x4 gz4 = {0, 0, 0, 0};          // groups 2/3 unused (2-D tensor)
        i32x8 gz8 = {0, 0, 0, 0, 0, 0, 0, 0};
        __builtin_amdgcn_tensor_load_to_lds(g0, g1, gz4, gz4, gz8, 0);
        __builtin_amdgcn_s_wait_tensorcnt(0);
    }
#else
    for (int i = threadIdx.x; i < STRIP * C_IN / 8; i += blockDim.x)
        ((v8bf*)lwt)[i] = ((const v8bf*)(Wt + (size_t)ng * STRIP * C_IN))[i];
#endif
    __syncthreads();

    const int mt = rb * 4 + wv;            // this wave's 16-row tile
    if (mt * 16 >= M) return;              // wave-uniform; EXEC all-ones below

    const int hf = lane >> 4;              // K-half selector
    const int qq = lane & 15;              // row (A) / col (B)

    const __bf16* arow = A + (size_t)(mt * 16 + qq) * C_IN;

    v8f c[4];
#pragma unroll
    for (int t = 0; t < 4; ++t) c[t] = (v8f){};

#pragma unroll
    for (int k0 = 0; k0 < C_IN; k0 += 32) {
        // A fragment from global: loaded once, reused for all 4 column tiles
        v8bf a0 = *(const v8bf*)(arow + k0 + hf * 8);
        v8bf a1 = *(const v8bf*)(arow + k0 + 16 + hf * 8);
        v16bf a;
#pragma unroll
        for (int i = 0; i < 8; ++i) { a[i] = a0[i]; a[i + 8] = a1[i]; }

#pragma unroll
        for (int t = 0; t < 4; ++t) {
            const __bf16* bp = lwt + (t * 16 + qq) * C_IN + k0 + hf * 8;
            v8bf b0 = *(const v8bf*)(bp);       // ds_load_b128
            v8bf b1 = *(const v8bf*)(bp + 16);
            v16bf b;
#pragma unroll
            for (int i = 0; i < 8; ++i) { b[i] = b0[i]; b[i + 8] = b1[i]; }
            c[t] = __builtin_amdgcn_wmma_f32_16x16x32_bf16(
                       false, a, false, b, (short)0, c[t], false, false);
        }
    }

#pragma unroll
    for (int t = 0; t < 4; ++t) {
        const int coln = ng * STRIP + t * 16 + qq;
        const float bvv = bias[coln];
#pragma unroll
        for (int r = 0; r < 8; ++r) {
            const int row = mt * 16 + hf * 8 + r;
            store_out(C + (size_t)row * ncols + coln, c[t][r] + bvv);
        }
    }
}

// ---- fp32 -> bf16 bulk convert ----
__global__ void cvt_f32_bf16(const float* __restrict__ in, __bf16* __restrict__ out, int n)
{
    int i = blockIdx.x * blockDim.x + threadIdx.x;
    if (i < n) out[i] = (__bf16)in[i];
}

// ---- transpose-convert W[K,Ncol] fp32 -> Wt[Ncol,K] bf16 ----
__global__ void cvt_transpose(const float* __restrict__ W, __bf16* __restrict__ Wt,
                              int K, int Ncol)
{
    int i = blockIdx.x * blockDim.x + threadIdx.x;
    if (i < K * Ncol) {
        int k = i % K, n = i / K;
        Wt[(size_t)n * K + k] = (__bf16)W[(size_t)k * Ncol + n];
    }
}

__global__ void init_stats(unsigned* maxenc, float* sums)
{
    int i = threadIdx.x;
    if (i < HEADS) { maxenc[i] = 0u; sums[i] = 0.f; }
}

__global__ void zero_f32(float* p, int n)
{
    int i = blockIdx.x * blockDim.x + threadIdx.x;
    if (i < n) p[i] = 0.f;
}

__device__ __forceinline__ unsigned enc_f32(float f)
{
    unsigned u = __float_as_uint(f);
    return (u & 0x80000000u) ? ~u : (u | 0x80000000u);
}
__device__ __forceinline__ float dec_f32(unsigned key)
{
    unsigned u = (key & 0x80000000u) ? (key & 0x7FFFFFFFu) : ~key;
    return __uint_as_float(u);
}

// ------------------------------------------------------------------
// One wave per edge: lanes split as head = lane>>3, sub = lane&7
// (8 lanes x 8 elems = D=64 per head). Dot(q[src],k[dst]) / sqrt(64)
// + geometric bias; block-reduced running max via encoded-uint atomics.
// ------------------------------------------------------------------
__global__ void edge_scores(const __bf16* __restrict__ qb, const __bf16* __restrict__ kb,
                            const float* __restrict__ pos, const int* __restrict__ ei,
                            const float* __restrict__ Wg, const float* __restrict__ bg,
                            float* __restrict__ scores, unsigned* __restrict__ maxenc, int E)
{
    __shared__ unsigned smax[HEADS];
    if (threadIdx.x < HEADS) smax[threadIdx.x] = 0u;
    __syncthreads();

    const int wid  = blockIdx.x * (blockDim.x >> 5) + (threadIdx.x >> 5);
    const int lane = threadIdx.x & 31;
    if (wid < E) {
        const int src = ei[wid];
        const int dst = ei[E + wid];
        const int h = lane >> 3, sub = lane & 7;

        const __bf16* qp = qb + (size_t)src * HD + h * DHEAD + sub * 8;
        const __bf16* kp = kb + (size_t)dst * HD + h * DHEAD + sub * 8;
        v8bf qv = *(const v8bf*)qp;
        v8bf kv = *(const v8bf*)kp;
        float acc = 0.f;
#pragma unroll
        for (int i = 0; i < 8; ++i) acc += (float)qv[i] * (float)kv[i];
        acc += __shfl_xor(acc, 1, 32);
        acc += __shfl_xor(acc, 2, 32);
        acc += __shfl_xor(acc, 4, 32);

        if (sub == 0) {
            float ex = pos[(size_t)dst * 3 + 0] - pos[(size_t)src * 3 + 0];
            float ey = pos[(size_t)dst * 3 + 1] - pos[(size_t)src * 3 + 1];
            float ez = pos[(size_t)dst * 3 + 2] - pos[(size_t)src * 3 + 2];
            float dist = sqrtf(ex * ex + ey * ey + ez * ez);
            float inv  = 1.f / (dist + 1e-8f);
            float geom = dist * Wg[0 * HEADS + h] + (ex * inv) * Wg[1 * HEADS + h] +
                         (ey * inv) * Wg[2 * HEADS + h] + (ez * inv) * Wg[3 * HEADS + h] +
                         bg[h];
            float s = acc * 0.125f + geom;   // 1/sqrt(64)
            scores[(size_t)wid * HEADS + h] = s;
            atomicMax(&smax[h], enc_f32(s));
        }
    }
    __syncthreads();
    if (threadIdx.x < HEADS) atomicMax(&maxenc[threadIdx.x], smax[threadIdx.x]);
}

// ---- pass 2: scores -> exp(scores - max); accumulate per-head sums ----
__global__ void softmax_expsum(float* __restrict__ scores,
                               const unsigned* __restrict__ maxenc,
                               float* __restrict__ sums, int total)
{
    __shared__ float ssum[HEADS];
    if (threadIdx.x < HEADS) ssum[threadIdx.x] = 0.f;
    __syncthreads();
    int i = blockIdx.x * blockDim.x + threadIdx.x;
    if (i < total) {
        int h = i & (HEADS - 1);
        float e = __expf(scores[i] - dec_f32(maxenc[h]));
        scores[i] = e;
        atomicAdd(&ssum[h], e);
    }
    __syncthreads();
    if (threadIdx.x < HEADS) atomicAdd(&sums[threadIdx.x], ssum[threadIdx.x]);
}

// ---- scalar scatter: coef[dst,h] += exp-score (softmax factored out of v) ----
__global__ void scatter_coef(const float* __restrict__ scores, const int* __restrict__ ei,
                             float* __restrict__ coef, int E)
{
    int i = blockIdx.x * blockDim.x + threadIdx.x;
    if (i < E * HEADS) {
        int e = i >> 2, h = i & (HEADS - 1);
        atomicAdd(&coef[(size_t)ei[E + e] * HEADS + h], scores[i]);
    }
}

// ---- h[n,c] = v[n,c] * coef[n, c/64] / sum[c/64]  (bf16 for final GEMM) ----
__global__ void mul_coef(const __bf16* __restrict__ vb, const float* __restrict__ coef,
                         const float* __restrict__ sums, __bf16* __restrict__ hb, int total)
{
    int i = blockIdx.x * blockDim.x + threadIdx.x;
    if (i < total) {
        int n = i >> 8, c = i & 255, h = c >> 6;
        hb[i] = (__bf16)((float)vb[i] * coef[(size_t)n * HEADS + h] * (1.f / sums[h]));
    }
}

extern "C" void kernel_launch(void* const* d_in, const int* in_sizes, int n_in,
                              void* d_out, int out_size, void* d_ws, size_t ws_size,
                              hipStream_t stream)
{
    const float* x   = (const float*)d_in[0];
    const int*   ei  = (const int*)d_in[1];
    const float* pos = (const float*)d_in[2];
    const float* Wq  = (const float*)d_in[3];
    const float* bq  = (const float*)d_in[4];
    const float* Wk  = (const float*)d_in[5];
    const float* bk  = (const float*)d_in[6];
    const float* Wv  = (const float*)d_in[7];
    const float* bv  = (const float*)d_in[8];
    const float* Wg  = (const float*)d_in[9];
    const float* bg  = (const float*)d_in[10];
    const float* Wo  = (const float*)d_in[11];
    const float* bo  = (const float*)d_in[12];

    const int N = in_sizes[0] / C_IN;   // 50000
    const int E = in_sizes[1] / 2;      // 800000
    float* out = (float*)d_out;         // [N, 64]

    // ---- workspace carve-out ----
    char* w = (char*)d_ws;
    auto carve = [&](size_t bytes) {
        char* p = w;
        w += (bytes + 255) & ~(size_t)255;
        return p;
    };
    __bf16* xb  = (__bf16*)carve((size_t)N * C_IN * 2);
    __bf16* qb  = (__bf16*)carve((size_t)N * HD * 2);
    __bf16* kb  = (__bf16*)carve((size_t)N * HD * 2);
    __bf16* vbv = (__bf16*)carve((size_t)N * HD * 2);
    __bf16* hb  = (__bf16*)carve((size_t)N * HD * 2);
    __bf16* wqt = (__bf16*)carve((size_t)HD * C_IN * 2);
    __bf16* wkt = (__bf16*)carve((size_t)HD * C_IN * 2);
    __bf16* wvt = (__bf16*)carve((size_t)HD * C_IN * 2);
    __bf16* wot = (__bf16*)carve((size_t)DHEAD * HD * 2);
    float*  scores = (float*)carve((size_t)E * HEADS * 4);
    float*  coef   = (float*)carve((size_t)N * HEADS * 4);
    unsigned* maxenc = (unsigned*)carve(HEADS * 4);
    float*    sums   = (float*)carve(HEADS * 4);

    const int T = 256;
    auto blk = [](int n, int t) { return (n + t - 1) / t; };

    // 0) init scalars + zero coefficient accumulator
    init_stats<<<1, 32, 0, stream>>>(maxenc, sums);
    zero_f32<<<blk(N * HEADS, T), T, 0, stream>>>(coef, N * HEADS);

    // 1) precision conversion / weight transposes
    cvt_f32_bf16<<<blk(N * C_IN, T), T, 0, stream>>>(x, xb, N * C_IN);
    cvt_transpose<<<blk(C_IN * HD, T), T, 0, stream>>>(Wq, wqt, C_IN, HD);
    cvt_transpose<<<blk(C_IN * HD, T), T, 0, stream>>>(Wk, wkt, C_IN, HD);
    cvt_transpose<<<blk(C_IN * HD, T), T, 0, stream>>>(Wv, wvt, C_IN, HD);
    cvt_transpose<<<blk(C_IN * DHEAD, T), T, 0, stream>>>(Wo, wot, C_IN, DHEAD);

    // 2) Q/K/V projections via WMMA. Grid = row-blocks x column strips;
    //    each block TDM-stages a 32KB weight strip to LDS shared by 4 waves.
    {
        int rowblocks = blk(N / 16, 4);
        int blocks = rowblocks * (HD / STRIP);
        gemm_bf16_wmma<__bf16><<<blocks, 128, 0, stream>>>(xb, wqt, bq, qb, N, HD);
        gemm_bf16_wmma<__bf16><<<blocks, 128, 0, stream>>>(xb, wkt, bk, kb, N, HD);
        gemm_bf16_wmma<__bf16><<<blocks, 128, 0, stream>>>(xb, wvt, bv, vbv, N, HD);
    }

    // 3) per-edge attention scores + global per-head max
    edge_scores<<<blk(E, 8), 256, 0, stream>>>(qb, kb, pos, ei, Wg, bg,
                                               scores, maxenc, E);

    // 4) exp + per-head sums (global softmax over all edges, per head)
    softmax_expsum<<<blk(E * HEADS, T), T, 0, stream>>>(scores, maxenc, sums, E * HEADS);

    // 5) scalar coefficient scatter:  coef[n,h] = sum_{e: dst=n} exp-score
    scatter_coef<<<blk(E * HEADS, T), T, 0, stream>>>(scores, ei, coef, E);

    // 6) h = v * coef / sum  (factored softmax-weighted segment sum)
    mul_coef<<<blk(N * HD, T), T, 0, stream>>>(vbv, coef, sums, hb, N * HD);

    // 7) output projection via WMMA -> d_out [N, 64] fp32
    {
        int rowblocks = blk(N / 16, 4);
        int blocks = rowblocks * (DHEAD / STRIP);
        gemm_bf16_wmma<float><<<blocks, 128, 0, stream>>>(hb, wot, bo, out, N, DHEAD);
    }
}
